// ActionExtractionHypersphericalResNet_23965917511977
// MI455X (gfx1250) — matrix operand order repaired and checked by reading, not compiled, not run on hardware
//
#include <hip/hip_runtime.h>

typedef _Float16 h4  __attribute__((ext_vector_type(4)));
typedef _Float16 h8  __attribute__((ext_vector_type(8)));
typedef _Float16 h16 __attribute__((ext_vector_type(16)));
typedef float    f8  __attribute__((ext_vector_type(8)));

// ---- problem sizes (fixed by the reference) ----
constexpr int FDIM = 512;   // F
constexpr int HDIM = 512;   // H
constexpr int TTR  = 4;     // T trials

// ---- tiling ----
constexpr int MB  = 128;    // rows per block (8 waves x 16 rows)
constexpr int NP  = 32;     // W2 N-panel width (double-buffered)

// f16-element strides (padded to break 64-bank conflicts)
constexpr int KS512 = 520;
constexpr int KS32  = 40;

constexpr int PANEL_B      = NP * KS512 * 2;   // 33280 B per W2 panel
constexpr int PANEL_CHUNKS = PANEL_B / 16;     // 2080 b128 transfers

// ---- LDS layout (bytes) ----
constexpr int OFF_X   = 0;                        // h tile / x1 : [128][520] f16
constexpr int OFF_W   = OFF_X   + MB * KS512 * 2; // 2x W2 panel bufs (also WmuT / W1T staging)
constexpr int OFF_W3  = OFF_W   + 2 * PANEL_B;    // W3T persistent: [32][520] f16
constexpr int OFF_X2P = OFF_W3  + 32 * KS512 * 2; // x2 panel: [128][40] f16
constexpr int OFF_Z   = OFF_X2P + MB * KS32 * 2;  // z: [128][40] f16
constexpr int OFF_MU  = OFF_Z   + MB * KS32 * 2;  // mu: [128][32] f32
constexpr int OFF_X3  = OFF_MU  + MB * 32 * 4;    // x3: [128][32] f32
constexpr int OFF_B   = OFF_X3  + MB * 32 * 4;    // biases/W4/Wk f32 pool
constexpr int SMEM_TOTAL = OFF_B + 7680;          // = 293888 B (< 320 KB WGP LDS)

// bias pool f32 indices
constexpr int LB_B1 = 0, LB_B2 = 512, LB_B3 = 1024, LB_B4 = 1056,
              LB_BMU = 1064, LB_BK = 1096, LB_W4 = 1104, LB_WK = 1328;

// ---- workspace layout (f16 elements) ----
constexpr int WS_WMU = 0;                       // [32][520]
constexpr int WS_W1  = WS_WMU + 32 * KS512;     // [512][40]
constexpr int WS_W2  = WS_W1  + 512 * KS32;     // [512][520]
constexpr int WS_W3  = WS_W2  + 512 * KS512;    // [32][520]  (total 640,000 B of ws)

#define WMMA_F16(acc, a, b) \
  acc = __builtin_amdgcn_wmma_f32_16x16x32_f16(false, (a), false, (b), (short)0, (acc), false, false)

static __device__ __forceinline__ h16 ld_pair(const char* p, int off2) {
  union { h16 v; h8 q[2]; } u;
  u.q[0] = *(const h8*)(p);
  u.q[1] = *(const h8*)(p + off2);
  return u.v;
}

// gfx1250 async global -> LDS copy, 16 B per lane, tracked by ASYNCcnt
static __device__ __forceinline__ void async_g2l_b128(unsigned ldsAddr, unsigned gOff,
                                                      const void* sbase) {
  asm volatile("global_load_async_to_lds_b128 %0, %1, %2"
               :: "v"(ldsAddr), "v"(gOff), "s"(sbase)
               : "memory");
}
static __device__ __forceinline__ void wait_async0() {
#if defined(__has_builtin)
#if __has_builtin(__builtin_amdgcn_s_wait_asynccnt)
  __builtin_amdgcn_s_wait_asynccnt(0);
  return;
#endif
#endif
  asm volatile("s_wait_asynccnt 0" ::: "memory");
}

// Convert fp32 [K][N] row-major -> f16 [N][kstride] (transposed, padded)
__global__ void prep_t(const float* __restrict__ src, _Float16* __restrict__ dst,
                       int K, int N, int kstride) {
  int i = blockIdx.x * 256 + threadIdx.x;
  if (i >= K * N) return;
  int k = i / N, n = i % N;
  dst[n * kstride + k] = (_Float16)src[i];
}

__global__ __launch_bounds__(256, 1)
void fused_vmf_mlp(const float* __restrict__ hg, const float* __restrict__ vg,
                   const float* __restrict__ zug, const float* __restrict__ uug,
                   const float* __restrict__ bmu, const float* __restrict__ Wk,
                   const float* __restrict__ bk,  const float* __restrict__ b1,
                   const float* __restrict__ b2,  const float* __restrict__ b3,
                   const float* __restrict__ b4,  const float* __restrict__ W4,
                   const _Float16* __restrict__ wsf,
                   float* __restrict__ out_o, float* __restrict__ out_mu,
                   float* __restrict__ out_k, int Btot) {
  extern __shared__ __align__(16) char smem[];
  const int tid   = threadIdx.x;
  const int lane  = tid & 31;
  const int wv    = tid >> 5;
  const int lhalf = lane & 15;
  const int hi    = lane >> 4;
  const int row0  = blockIdx.x * MB;
  const int wrow0 = wv * 16;
  float* lb = (float*)(smem + OFF_B);

  // ---------- phase 0: fills ----------
  for (int i = tid; i < MB * FDIM / 4; i += 256) {     // h tile -> f16 LDS
    int r = i >> 7;
    int k = (i & 127) << 2;
    float4 t4 = *(const float4*)(hg + (size_t)(row0 + r) * FDIM + k);
    h4 q = { (_Float16)t4.x, (_Float16)t4.y, (_Float16)t4.z, (_Float16)t4.w };
    *(h4*)(smem + OFF_X + r * (KS512 * 2) + k * 2) = q;
  }
  { // WmuT -> lds_w ; W3T -> lds_w3
    const uint4* s0 = (const uint4*)(wsf + WS_WMU);
    uint4* d0 = (uint4*)(smem + OFF_W);
    const uint4* s1 = (const uint4*)(wsf + WS_W3);
    uint4* d1 = (uint4*)(smem + OFF_W3);
    for (int i = tid; i < 2080; i += 256) { d0[i] = s0[i]; d1[i] = s1[i]; }
  }
  for (int i = tid; i < 512; i += 256) {
    lb[LB_B1 + i] = b1[i]; lb[LB_B2 + i] = b2[i]; lb[LB_WK + i] = Wk[i];
  }
  if (tid < 32)  { lb[LB_B3 + tid] = b3[tid]; lb[LB_BMU + tid] = bmu[tid]; }
  if (tid < 7)   lb[LB_B4 + tid] = b4[tid];
  if (tid < 224) lb[LB_W4 + tid] = W4[tid];
  if (tid == 0)  lb[LB_BK] = bk[0];
  __syncthreads();

  // ---------- stage A: mu = normalize(h @ W_mu + b_mu), kappa ----------
  f8 accT0 = {}, accT1 = {};
#pragma unroll
  for (int kc = 0; kc < 16; ++kc) {
    h16 a  = ld_pair(smem + OFF_X + (wrow0 + lhalf) * (KS512 * 2) + kc * 64 + hi * 16, 32);
    h16 bA = ld_pair(smem + OFF_W + (lhalf)        * (KS512 * 2) + kc * 64 + hi * 16, 32);
    h16 bB = ld_pair(smem + OFF_W + (16 + lhalf)   * (KS512 * 2) + kc * 64 + hi * 16, 32);
    WMMA_F16(accT0, a, bA);
    WMMA_F16(accT1, a, bB);
  }
  {
    float bm0 = lb[LB_BMU + lhalf], bm1 = lb[LB_BMU + 16 + lhalf];
#pragma unroll
    for (int vgi = 0; vgi < 8; ++vgi) {
      float r0 = accT0[vgi] + bm0, r1 = accT1[vgi] + bm1;
      float sq = r0 * r0 + r1 * r1;
      sq += __shfl_xor(sq, 1); sq += __shfl_xor(sq, 2);
      sq += __shfl_xor(sq, 4); sq += __shfl_xor(sq, 8);
      float inv = 1.0f / fmaxf(sqrtf(sq), 1e-12f);
      r0 *= inv; r1 *= inv;
      int lr = wrow0 + vgi + hi * 8;
      float* mup = (float*)(smem + OFF_MU) + lr * 32;
      mup[lhalf] = r0; mup[16 + lhalf] = r1;
      size_t gr = (size_t)(row0 + lr);
      out_mu[gr * 32 + lhalf] = r0; out_mu[gr * 32 + 16 + lhalf] = r1;
    }
  }
  float kap;
  {
    float ka = 0.0f;
    const char* xrow = smem + OFF_X + (wrow0 + lhalf) * (KS512 * 2) + hi * 512;
    for (int j = 0; j < 32; ++j) {
      h8 hv = *(const h8*)(xrow + j * 16);
#pragma unroll
      for (int i = 0; i < 8; ++i)
        ka += (float)hv[i] * lb[LB_WK + hi * 256 + j * 8 + i];
    }
    ka += __shfl_xor(ka, 16);
    ka += lb[LB_BK];
    kap = fmaxf(ka, 0.0f) + log1pf(expf(-fabsf(ka))) + 1.0f;  // softplus + 1
    if (!hi) out_k[row0 + wrow0 + lhalf] = kap;
  }

  // ---------- stage B: vMF radial sample + Householder reflection -> z (f16) ----------
  {
    const int grow = row0 + wrow0 + lhalf;
    const int db = hi * 16;
    float vv[16], uv[16];
    const float* vp = vg + (size_t)grow * 32 + db;
#pragma unroll
    for (int q = 0; q < 4; ++q) {
      float4 t4 = *(const float4*)(vp + q * 4);
      vv[q * 4] = t4.x; vv[q * 4 + 1] = t4.y; vv[q * 4 + 2] = t4.z; vv[q * 4 + 3] = t4.w;
    }
    float ss = 0.0f;
#pragma unroll
    for (int i = 0; i < 16; ++i) ss += vv[i] * vv[i];
    ss += __shfl_xor(ss, 16);
    float inv = 1.0f / fmaxf(sqrtf(ss), 1e-12f);
    float ssu = 0.0f;
    const float* mup = (float*)(smem + OFF_MU) + (wrow0 + lhalf) * 32 + db;
#pragma unroll
    for (int i = 0; i < 16; ++i) {
      vv[i] *= inv;
      float u = vv[i] - mup[i];
      uv[i] = u; ssu += u * u;
    }
    ssu += __shfl_xor(ssu, 16);
    float invu = 1.0f / fmaxf(sqrtf(ssu), 1e-12f);
    float dot = 0.0f;
#pragma unroll
    for (int i = 0; i < 16; ++i) { uv[i] *= invu; dot += vv[i] * uv[i]; }
    dot += __shfl_xor(dot, 16);
    float e = expf(-2.0f * kap);
    float wsel = 0.0f; bool done = false;
#pragma unroll
    for (int t = 0; t < TTR; ++t) {
      float zr = zug[(size_t)t * Btot + grow];
      float ur = uug[(size_t)t * Btot + grow];
      float zz = zr * (1.0f - e) + e;
      float wt = 1.0f - logf(zz) / kap;
      bool acc = (logf(ur) <= 29.0f * logf(wt) + kap * wt);  // (D-3)=29
      if (acc && !done) wsel = wt;
      done = done || acc;
    }
    _Float16 hz[16];
#pragma unroll
    for (int i = 0; i < 16; ++i)
      hz[i] = (_Float16)((vv[i] - 2.0f * dot * uv[i]) * wsel);
    char* zp = smem + OFF_Z + (wrow0 + lhalf) * (KS32 * 2) + hi * 32;
    *(h8*)zp = *(h8*)&hz[0];
    *(h8*)(zp + 16) = *(h8*)&hz[8];
  }

  // ---------- layer 1: x1 = relu(z @ W1 + b1) ----------
  __syncthreads();  // everyone done with WmuT in lds_w
  {
    const uint4* s0 = (const uint4*)(wsf + WS_W1);
    uint4* d0 = (uint4*)(smem + OFF_W);
    for (int i = tid; i < 2560; i += 256) d0[i] = s0[i];
  }
  __syncthreads();
  {
    h16 az = ld_pair(smem + OFF_Z + (wrow0 + lhalf) * (KS32 * 2) + hi * 16, 32);
    for (int nt = 0; nt < 32; ++nt) {
      f8 acc = {};
      h16 b = ld_pair(smem + OFF_W + (nt * 16 + lhalf) * (KS32 * 2) + hi * 16, 32);
      WMMA_F16(acc, az, b);
      float bias = lb[LB_B1 + nt * 16 + lhalf];
#pragma unroll
      for (int vgi = 0; vgi < 8; ++vgi) {
        float r = fmaxf(acc[vgi] + bias, 0.0f);
        *(_Float16*)(smem + OFF_X + (wrow0 + vgi + hi * 8) * (KS512 * 2) +
                     (nt * 16 + lhalf) * 2) = (_Float16)r;
      }
    }
  }

  // hoist this wave's x1 A-fragments (16 K-chunks) into registers for all of layer 2
  h16 afr[16];
#pragma unroll
  for (int kc = 0; kc < 16; ++kc)
    afr[kc] = ld_pair(smem + OFF_X + (wrow0 + lhalf) * (KS512 * 2) + kc * 64 + hi * 16, 32);

  __syncthreads();  // all waves done with W1T before panel 0 overwrites lds_w

  // prologue: async-copy W2 panel 0 into buffer 0
  for (int i = tid; i < PANEL_CHUNKS; i += 256) {
    unsigned lds = (unsigned)(unsigned long long)(smem + OFF_W + i * 16);
    async_g2l_b128(lds, (unsigned)(WS_W2 * 2) + (unsigned)(i * 16), (const void*)wsf);
  }

  // ---------- layer 2 (16 double-buffered async panels) + fused layer-3 ----------
  f8 x3a0 = {}, x3a1 = {};
  for (int np = 0; np < HDIM / NP; ++np) {
    wait_async0();        // own async issues (panel np) complete
    __syncthreads();      // panel np visible block-wide; prev buffer free
    if (np + 1 < HDIM / NP) {
      const int buf = (np + 1) & 1;
      unsigned gbase = (unsigned)(WS_W2 * 2) + (unsigned)((np + 1) * PANEL_B);
      for (int i = tid; i < PANEL_CHUNKS; i += 256) {
        unsigned lds = (unsigned)(unsigned long long)(smem + OFF_W + buf * PANEL_B + i * 16);
        async_g2l_b128(lds, gbase + (unsigned)(i * 16), (const void*)wsf);
      }
    }
    const char* wbuf = smem + OFF_W + (np & 1) * PANEL_B;
    for (int nt = 0; nt < 2; ++nt) {
      f8 acc = {};
#pragma unroll
      for (int kc = 0; kc < 16; ++kc) {
        h16 b = ld_pair(wbuf + (nt * 16 + lhalf) * (KS512 * 2) + kc * 64 + hi * 16, 32);
        WMMA_F16(acc, afr[kc], b);
      }
      float bias = lb[LB_B2 + np * NP + nt * 16 + lhalf];
#pragma unroll
      for (int vgi = 0; vgi < 8; ++vgi) {
        float r = fmaxf(acc[vgi] + bias, 0.0f);   // relu(x2) panel
        *(_Float16*)(smem + OFF_X2P + (wrow0 + vgi + hi * 8) * (KS32 * 2) +
                     (nt * 16 + lhalf) * 2) = (_Float16)r;
      }
    }
    // x3 += relu(x2_panel) @ W3[panel_k_slice, :]   (K = 32 per panel)
    {
      h16 a = ld_pair(smem + OFF_X2P + (wrow0 + lhalf) * (KS32 * 2) + hi * 16, 32);
      h16 bA = ld_pair(smem + OFF_W3 + (lhalf)      * (KS512 * 2) + np * 64 + hi * 16, 32);
      h16 bB = ld_pair(smem + OFF_W3 + (16 + lhalf) * (KS512 * 2) + np * 64 + hi * 16, 32);
      WMMA_F16(x3a0, a, bA);
      WMMA_F16(x3a1, a, bB);
    }
  }

  // ---------- layer 3 finalize: relu + stash to LDS ----------
  {
    float b30 = lb[LB_B3 + lhalf], b31 = lb[LB_B3 + 16 + lhalf];
    float* x3p = (float*)(smem + OFF_X3);
#pragma unroll
    for (int vgi = 0; vgi < 8; ++vgi) {
      int lr = wrow0 + vgi + hi * 8;
      x3p[lr * 32 + lhalf]      = fmaxf(x3a0[vgi] + b30, 0.0f);
      x3p[lr * 32 + 16 + lhalf] = fmaxf(x3a1[vgi] + b31, 0.0f);
    }
  }

  // ---------- layer 4: out = x3 @ W4 + b4 (32x7 per row, VALU) ----------
  if (!hi) {
    const float* x3p = (float*)(smem + OFF_X3) + (wrow0 + lhalf) * 32;
    float s[7];
#pragma unroll
    for (int o = 0; o < 7; ++o) s[o] = lb[LB_B4 + o];
    for (int d = 0; d < 32; ++d) {
      float xd = x3p[d];
#pragma unroll
      for (int o = 0; o < 7; ++o) s[o] = fmaf(xd, lb[LB_W4 + d * 7 + o], s[o]);
    }
    size_t gr = (size_t)(row0 + wrow0 + lhalf);
#pragma unroll
    for (int o = 0; o < 7; ++o) out_o[gr * 7 + o] = s[o];
  }
}

extern "C" void kernel_launch(void* const* d_in, const int* in_sizes, int n_in,
                              void* d_out, int out_size, void* d_ws, size_t ws_size,
                              hipStream_t stream) {
  (void)n_in; (void)out_size; (void)ws_size;  // ws use = 640,000 B
  const float* h   = (const float*)d_in[0];
  const float* v   = (const float*)d_in[1];
  const float* zu  = (const float*)d_in[2];
  const float* uu  = (const float*)d_in[3];
  const float* Wmu = (const float*)d_in[4];
  const float* bmu = (const float*)d_in[5];
  const float* Wk  = (const float*)d_in[6];
  const float* bk  = (const float*)d_in[7];
  const float* W1  = (const float*)d_in[8];
  const float* b1  = (const float*)d_in[9];
  const float* W2  = (const float*)d_in[10];
  const float* b2  = (const float*)d_in[11];
  const float* W3  = (const float*)d_in[12];
  const float* b3  = (const float*)d_in[13];
  const float* W4  = (const float*)d_in[14];
  const float* b4  = (const float*)d_in[15];

  const int Btot = in_sizes[0] / FDIM;  // 131072
  _Float16* wsf = (_Float16*)d_ws;
  float* out = (float*)d_out;
  float* out_o  = out;
  float* out_mu = out + (size_t)Btot * 7;
  float* out_k  = out + (size_t)Btot * 39;

  prep_t<<<(512 * 32 + 255) / 256, 256, 0, stream>>>(Wmu, wsf + WS_WMU, 512, 32, KS512);
  prep_t<<<(32 * 512 + 255) / 256, 256, 0, stream>>>(W1, wsf + WS_W1, 32, 512, KS32);
  prep_t<<<(512 * 512 + 255) / 256, 256, 0, stream>>>(W2, wsf + WS_W2, 512, 512, KS512);
  prep_t<<<(512 * 32 + 255) / 256, 256, 0, stream>>>(W3, wsf + WS_W3, 512, 32, KS512);

  hipFuncSetAttribute((const void*)fused_vmf_mlp,
                      hipFuncAttributeMaxDynamicSharedMemorySize, SMEM_TOTAL);
  fused_vmf_mlp<<<Btot / MB, 256, SMEM_TOTAL, stream>>>(
      h, v, zu, uu, bmu, Wk, bk, b1, b2, b3, b4, W4, wsf, out_o, out_mu, out_k, Btot);
}
